// Down_7395933684151
// MI455X (gfx1250) — matrix operand order repaired and checked by reading, not compiled, not run on hardware
//
#include <hip/hip_runtime.h>
#include <math.h>

// ---------------------------------------------------------------------------
// MI455X / gfx1250. wave32, 4 waves per 128-thread block. Each wave computes
// a 32x32 f32 tile as 2x2 register-blocked v_wmma_f32_16x16x32_bf16 (4 WMMA
// per K-step from 8 global_load_b128 -> 2 loads per WMMA). All GEMM operands
// are materialized K-major in bf16 so fragment loads are contiguous 16B and
// the hot loops have no div/mod/branches.
// ---------------------------------------------------------------------------

typedef __attribute__((ext_vector_type(16))) __bf16 v16bf;
typedef __attribute__((ext_vector_type(8)))  float  v8f;
typedef __attribute__((ext_vector_type(4)))  unsigned int v4u;

#define WPB 4            // 32x32 tiles (waves) per block along N
#define TPB (WPB * 32)   // 128 threads = 4 wave32

struct PamGeom {
  int lofs[5];       // cumulative pixel counts per scale; lofs[4] = Lt
  int Hs[4], Ws[4];
  int refm_off[4];   // element offsets of per-scale [B,C2,Hs,Ws] chunks (f32)
  int base_off[4];   // element offsets of per-scale [B,C, Hs,Ws] chunks (f32)
};

__device__ __forceinline__ __bf16 f2bf(float f) {   // RNE f32 -> bf16
  union { float f; unsigned u; } in; in.f = f;
  unsigned r = in.u + 0x7FFFu + ((in.u >> 16) & 1u);
  union { unsigned short s; __bf16 b; } out; out.s = (unsigned short)(r >> 16);
  return out.b;
}

// Fragment layouts (ISA 7.12.2, 16-bit operands, wave32):
//  A (16x32 MxK): lane owns row m = lane&15; two contiguous 8-elem K-runs at
//    half and half+16, half = (lane&16) ? 8 : 0.
//  B (32x16 KxN): lane owns col n = lane&15; two 8-elem K-runs at half and half+16, as A.
union ABFrag { v16bf v; v4u q[2]; };

__device__ __forceinline__ void ld_a(ABFrag& f, const __bf16* p) {
  f.q[0] = *(const v4u*)(p);
  f.q[1] = *(const v4u*)(p + 16);
}
__device__ __forceinline__ void ld_b(ABFrag& f, const __bf16* p) {
  f.q[0] = *(const v4u*)(p);
  f.q[1] = *(const v4u*)(p + 16);
}
#define WMMA_BF16(A_, B_, C_) \
  __builtin_amdgcn_wmma_f32_16x16x32_bf16(false, (A_), false, (B_), (short)0, (C_), false, false)

// 2x2 register-blocked K-loop. aR0/aR1 = A rows m and m+16 (lane-half offset
// applied); bR0/bR1 = BT rows n and n+16.
__device__ __forceinline__ void gemm_2x2(const __bf16* __restrict__ aR0,
                                         const __bf16* __restrict__ aR1,
                                         const __bf16* __restrict__ bR0,
                                         const __bf16* __restrict__ bR1,
                                         int K, v8f& c00, v8f& c01, v8f& c10, v8f& c11) {
#pragma unroll 2
  for (int k0 = 0; k0 < K; k0 += 32) {
    __builtin_prefetch(aR0 + k0 + 256, 0, 0);   // global_prefetch_b8
    __builtin_prefetch(bR0 + k0 + 256, 0, 0);
    ABFrag a0, a1, b0, b1;
    ld_a(a0, aR0 + k0); ld_a(a1, aR1 + k0);
    ld_b(b0, bR0 + k0); ld_b(b1, bR1 + k0);
    c00 = WMMA_BF16(a0.v, b0.v, c00);
    c01 = WMMA_BF16(a0.v, b1.v, c01);
    c10 = WMMA_BF16(a1.v, b0.v, c10);
    c11 = WMMA_BF16(a1.v, b1.v, c11);
  }
}

// ---------------------------------------------------------------------------
// MaxPool2d(2)
__global__ void k_maxpool2(const float* __restrict__ in, float* __restrict__ out,
                           int BC, int Ho, int Wo) {
  int idx = blockIdx.x * blockDim.x + threadIdx.x;
  if (idx >= BC * Ho * Wo) return;
  int x = idx % Wo, y = (idx / Wo) % Ho, c = idx / (Wo * Ho);
  int Wi = 2 * Wo;
  const float* pl = in + (size_t)c * (2 * Ho) * Wi + (2 * y) * Wi + 2 * x;
  out[idx] = fmaxf(fmaxf(pl[0], pl[1]), fmaxf(pl[Wi], pl[Wi + 1]));
}

// flat f32 -> bf16 copy (weights)
__global__ void k_f2bf(const float* __restrict__ in, __bf16* __restrict__ out, int n) {
  int i = blockIdx.x * blockDim.x + threadIdx.x;
  if (i < n) out[i] = f2bf(in[i]);
}

// zero-fill bf16
__global__ void k_zero_bf(__bf16* __restrict__ out, int n) {
  int i = blockIdx.x * blockDim.x + threadIdx.x;
  if (i < n) out[i] = f2bf(0.f);
}

// NCHW f32 -> im2col bf16 [M = B*H*W, K = Ci*9], k = ci*9 + ky*3 + kx.
__global__ void k_im2col3(const float* __restrict__ in, __bf16* __restrict__ out,
                          int B, int Ci, int H, int W) {
  int HW = H * W, K = Ci * 9;
  long long idx = (long long)blockIdx.x * blockDim.x + threadIdx.x;
  if (idx >= (long long)B * HW * K) return;
  int k = (int)(idx % K); long long m = idx / K;
  int b = (int)(m / HW), p = (int)(m % HW);
  int ci = k / 9, r = k % 9;
  int yy = p / W + r / 3 - 1, xx = p % W + r % 3 - 1;
  float v = 0.f;
  if ((unsigned)yy < (unsigned)H && (unsigned)xx < (unsigned)W)
    v = in[((size_t)(b * Ci + ci) * H + yy) * W + xx];
  out[idx] = f2bf(v);
}

// NCHW f32 -> NHWC bf16 [M = B*HW, C]
__global__ void k_nhwc(const float* __restrict__ in, __bf16* __restrict__ out,
                       int B, int C, int HW) {
  long long idx = (long long)blockIdx.x * blockDim.x + threadIdx.x;
  if (idx >= (long long)B * HW * C) return;
  int c = (int)(idx % C); long long m = idx / C;
  int b = (int)(m / HW), p = (int)(m % HW);
  out[idx] = f2bf(in[(size_t)(b * C + c) * HW + p]);
}

// ---------------------------------------------------------------------------
// Conv GEMM: A [Mpad, K] bf16 (K-major) x Wt [Co, K] bf16. Epilogue: bias +
// act(0 none / 1 relu / 2 prelu) + optional residual, NCHW f32 store.
__global__ void k_gemm_conv_wmma(const __bf16* __restrict__ A, const __bf16* __restrict__ Wt,
                                 const float* __restrict__ bias, const float* __restrict__ alpha,
                                 const float* __restrict__ resid, float* __restrict__ out,
                                 int M, int K, int Co, int HW, int act) {
  int lane = threadIdx.x & 31;
  int n0 = (blockIdx.x * WPB + (threadIdx.x >> 5)) * 32;
  if (n0 >= Co) return;
  int m0 = blockIdx.y * 32;
  int ah = (lane & 16) ? 8 : 0, bh = (lane & 16) ? 8 : 0;
  const __bf16* aR0 = A + (size_t)(m0 + (lane & 15)) * K + ah;
  const __bf16* bR0 = Wt + (size_t)(n0 + (lane & 15)) * K + bh;
  v8f c00 = {}, c01 = {}, c10 = {}, c11 = {};
  gemm_2x2(aR0, aR0 + (size_t)16 * K, bR0, bR0 + (size_t)16 * K, K, c00, c01, c10, c11);
  float a = (act == 2) ? alpha[0] : 0.f;
  v8f accs[2][2] = {{c00, c01}, {c10, c11}};
#pragma unroll
  for (int ti = 0; ti < 2; ++ti)
#pragma unroll
    for (int tj = 0; tj < 2; ++tj) {
      int co = n0 + tj * 16 + (lane & 15);
      float bv = bias[co];
#pragma unroll
      for (int v = 0; v < 8; ++v) {
        int mi = m0 + ti * 16 + ((lane & 16) ? 8 : 0) + v;
        if (mi >= M) continue;
        float r = accs[ti][tj][v] + bv;
        if (act == 1) r = fmaxf(r, 0.f);
        else if (act == 2) r = (r >= 0.f) ? r : a * r;
        int b = mi / HW, p = mi % HW;
        size_t oi = (size_t)(b * Co + co) * HW + p;
        out[oi] = r + (resid ? resid[oi] : 0.f);
      }
    }
}

// ---------------------------------------------------------------------------
// Bilinear resize (half-pixel centers, edge clamp), per [B*C] plane.
__global__ void k_resize(const float* __restrict__ in, float* __restrict__ out,
                         int BC, int Hi, int Wi, int Ho, int Wo) {
  int idx = blockIdx.x * blockDim.x + threadIdx.x;
  if (idx >= BC * Ho * Wo) return;
  int x = idx % Wo, y = (idx / Wo) % Ho, c = idx / (Wo * Ho);
  float sy = (y + 0.5f) * (float)Hi / Ho - 0.5f;
  float sx = (x + 0.5f) * (float)Wi / Wo - 0.5f;
  int y0 = (int)floorf(sy), x0 = (int)floorf(sx);
  float fy = sy - y0, fx = sx - x0;
  int y0c = min(max(y0, 0), Hi - 1), y1c = min(max(y0 + 1, 0), Hi - 1);
  int x0c = min(max(x0, 0), Wi - 1), x1c = min(max(x0 + 1, 0), Wi - 1);
  const float* pl = in + (size_t)c * Hi * Wi;
  float v00 = pl[y0c * Wi + x0c], v01 = pl[y0c * Wi + x1c];
  float v10 = pl[y1c * Wi + x0c], v11 = pl[y1c * Wi + x1c];
  out[idx] = (1.f - fy) * ((1.f - fx) * v00 + fx * v01) +
             fy * ((1.f - fx) * v10 + fx * v11);
}

// ---------------------------------------------------------------------------
// xpT[(b*P + p)*576 + k] (bf16): query patches of match_base, K-major per
// pixel -> transposed B operand of the score GEMM.
__global__ void k_make_xpT(const float* __restrict__ mb, __bf16* __restrict__ xpT,
                           int B, int C2, int Hp, int Wp) {
  int HW = Hp * Wp, K = C2 * 9;
  long long idx = (long long)blockIdx.x * blockDim.x + threadIdx.x;
  if (idx >= (long long)B * HW * K) return;
  int k = (int)(idx % K); long long m = idx / K;
  int b = (int)(m / HW), p = (int)(m % HW);
  int c = k / 9, r = k % 9;
  int yy = p / Wp + r / 3 - 1, xx = p % Wp + r % 3 - 1;
  float v = 0.f;
  if ((unsigned)yy < (unsigned)Hp && (unsigned)xx < (unsigned)Wp)
    v = mb[((size_t)(b * C2 + c) * Hp + yy) * Wp + xx];
  xpT[idx] = f2bf(v);
}

// ---------------------------------------------------------------------------
// wn[(b*Ltp + l)*576 + k] (bf16): L2-normalized conv_match patches.
// One wave per (b,l); wave32 reduction via shfl_xor.
__global__ void k_make_wn(const float* __restrict__ refm_all, __bf16* __restrict__ wn,
                          PamGeom g, int B, int C2, int Ltp) {
  int row = blockIdx.x * 8 + (threadIdx.x >> 5);
  int lane = threadIdx.x & 31;
  int Lt = g.lofs[4];
  if (row >= B * Lt) return;
  int b = row / Lt, l = row % Lt;
  int s = 0;
  while (s < 3 && l >= g.lofs[s + 1]) ++s;
  int li = l - g.lofs[s], Hs = g.Hs[s], Ws = g.Ws[s];
  int ys = li / Ws, xs = li % Ws;
  float vals[18];
  float ss = 0.f;
#pragma unroll
  for (int i = 0; i < 18; ++i) {
    int k = lane + 32 * i;
    int c = k / 9, r = k % 9;
    int yy = ys + r / 3 - 1, xx = xs + r % 3 - 1;
    float v = 0.f;
    if ((unsigned)yy < (unsigned)Hs && (unsigned)xx < (unsigned)Ws)
      v = refm_all[g.refm_off[s] + ((size_t)(b * C2 + c) * Hs + yy) * Ws + xx];
    vals[i] = v; ss += v * v;
  }
  for (int o = 16; o > 0; o >>= 1) ss += __shfl_xor(ss, o, 32);
  float inv = 1.f / fmaxf(sqrtf(ss), 1e-4f);
  size_t base = (size_t)(b * Ltp + l) * 576;
#pragma unroll
  for (int i = 0; i < 18; ++i) wn[base + lane + 32 * i] = f2bf(vals[i] * inv);
}

// ---------------------------------------------------------------------------
// scores[b, l, p] = 10 * <wn[b,l,:], xpT[b,p,:]>   (M=Lt pad Ltp, N=P, K=576)
__global__ void k_gemm_scores_wmma(const __bf16* __restrict__ Awn, const __bf16* __restrict__ XpT,
                                   float* __restrict__ C, int Lt, int Ltp, int P, int K,
                                   float scale) {
  int bz = blockIdx.z;
  const __bf16* A  = Awn + (size_t)bz * Ltp * K;
  const __bf16* Bt = XpT + (size_t)bz * P * K;
  float* Cb = C + (size_t)bz * Lt * P;
  int lane = threadIdx.x & 31;
  int n0 = (blockIdx.x * WPB + (threadIdx.x >> 5)) * 32;
  int m0 = blockIdx.y * 32;
  int ah = (lane & 16) ? 8 : 0, bh = (lane & 16) ? 8 : 0;
  const __bf16* aR0 = A + (size_t)(m0 + (lane & 15)) * K + ah;
  const __bf16* bR0 = Bt + (size_t)(n0 + (lane & 15)) * K + bh;
  v8f c00 = {}, c01 = {}, c10 = {}, c11 = {};
  gemm_2x2(aR0, aR0 + (size_t)16 * K, bR0, bR0 + (size_t)16 * K, K, c00, c01, c10, c11);
  v8f accs[2][2] = {{c00, c01}, {c10, c11}};
#pragma unroll
  for (int ti = 0; ti < 2; ++ti)
#pragma unroll
    for (int tj = 0; tj < 2; ++tj) {
      int n = n0 + tj * 16 + (lane & 15);
#pragma unroll
      for (int v = 0; v < 8; ++v) {
        int mi = m0 + ti * 16 + ((lane & 16) ? 8 : 0) + v;
        if (mi < Lt) Cb[(size_t)mi * P + n] = accs[ti][tj][v] * scale;
      }
    }
}

// ---------------------------------------------------------------------------
// Online softmax over l per (b,p). Reads scores [b][l][p] (coalesced across
// threads), writes attT [b][p][Ltp] rows (bf16, K-major), pad zero-filled.
__global__ void k_softmax(const float* __restrict__ scores, __bf16* __restrict__ attT,
                          int B, int Lt, int Ltp, int P) {
  int tid = blockIdx.x * blockDim.x + threadIdx.x;
  if (tid >= B * P) return;
  int b = tid / P, p = tid % P;
  const float* col = scores + (size_t)b * Lt * P + p;
  float mx = -3.0e38f, d = 0.f;
  for (int l = 0; l < Lt; ++l) {
    float s = col[(size_t)l * P];
    float nm = fmaxf(mx, s);
    d = d * __expf(mx - nm) + __expf(s - nm);
    mx = nm;
  }
  float inv = 1.f / d;
  __bf16* row = attT + ((size_t)b * P + p) * Ltp;
  for (int l = 0; l < Lt; ++l)
    row[l] = f2bf(__expf(col[(size_t)l * P] - mx) * inv);
  __bf16 z0 = f2bf(0.f);
  for (int l = Lt; l < Ltp; ++l) row[l] = z0;
}

// ---------------------------------------------------------------------------
// rwT[((b*9 + r)*Co + co)*Ltp + l] (bf16): flipped conv_assembly patches,
// N-major rows of length Ltp (zero-padded) -> BT operand of the tconv GEMMs.
__global__ void k_make_rwT(const float* __restrict__ base_all, __bf16* __restrict__ rwT,
                           PamGeom g, int B, int C, int Ltp) {
  int Lt = g.lofs[4];
  long long idx = (long long)blockIdx.x * blockDim.x + threadIdx.x;
  if (idx >= (long long)B * 9 * C * Ltp) return;
  int l = (int)(idx % Ltp); long long t = idx / Ltp;
  int co = (int)(t % C); long long t2 = t / C;
  int r = (int)(t2 % 9); int b = (int)(t2 / 9);
  float v = 0.f;
  if (l < Lt) {
    int s = 0;
    while (s < 3 && l >= g.lofs[s + 1]) ++s;
    int li = l - g.lofs[s], Hs = g.Hs[s], Ws = g.Ws[s];
    int yy = li / Ws + 1 - r / 3, xx = li % Ws + 1 - r % 3;
    if ((unsigned)yy < (unsigned)Hs && (unsigned)xx < (unsigned)Ws)
      v = base_all[g.base_off[s] + ((size_t)(b * C + co) * Hs + yy) * Ws + xx];
  }
  rwT[idx] = f2bf(v);
}

// ---------------------------------------------------------------------------
// tconv: out[b,co,p] = resid + 0.25 * sum_r sum_l attT[b, shift_r(p), l] *
// rwT[b,r,co,l]. Nine accumulating K=Ltp GEMMs; border lanes read a zeroed
// dummy row so the inner loop is branch-free.
__global__ void k_tconv_wmma(const __bf16* __restrict__ attT, const __bf16* __restrict__ rwT,
                             const __bf16* __restrict__ zrow, const float* __restrict__ resid,
                             float* __restrict__ out, int Ltp, int Co, int Hp, int Wp) {
  int lane = threadIdx.x & 31;
  int n0 = (blockIdx.x * WPB + (threadIdx.x >> 5)) * 32;
  int m0 = blockIdx.y * 32;
  int bz = blockIdx.z;
  if (n0 >= Co) return;
  int P = Hp * Wp;
  int mp0 = m0 + (lane & 15);         // lane's A pixels (rows m, m+16)
  int mp1 = mp0 + 16;
  int py0 = mp0 / Wp, px0 = mp0 % Wp;
  int py1 = mp1 / Wp, px1 = mp1 % Wp;
  int ah = (lane & 16) ? 8 : 0, bh = (lane & 16) ? 8 : 0;
  int nb = n0 + (lane & 15);          // lane's B rows (co, co+16)
  const __bf16* attb = attT + (size_t)bz * P * Ltp;
  v8f c00 = {}, c01 = {}, c10 = {}, c11 = {};
  for (int r = 0; r < 9; ++r) {
    int dy = r / 3 - 1, dx = r % 3 - 1;
    int y0 = py0 + dy, x0 = px0 + dx, y1 = py1 + dy, x1 = px1 + dx;
    const __bf16* aR0 =
        ((unsigned)y0 < (unsigned)Hp && (unsigned)x0 < (unsigned)Wp)
            ? attb + (size_t)(y0 * Wp + x0) * Ltp + ah : zrow + ah;
    const __bf16* aR1 =
        ((unsigned)y1 < (unsigned)Hp && (unsigned)x1 < (unsigned)Wp)
            ? attb + (size_t)(y1 * Wp + x1) * Ltp + ah : zrow + ah;
    const __bf16* bR0 = rwT + ((size_t)((bz * 9 + r) * Co + nb)) * Ltp + bh;
    gemm_2x2(aR0, aR1, bR0, bR0 + (size_t)16 * Ltp, Ltp, c00, c01, c10, c11);
  }
  v8f accs[2][2] = {{c00, c01}, {c10, c11}};
#pragma unroll
  for (int ti = 0; ti < 2; ++ti)
#pragma unroll
    for (int tj = 0; tj < 2; ++tj) {
      int co = n0 + tj * 16 + (lane & 15);
#pragma unroll
      for (int v = 0; v < 8; ++v) {
        int p = m0 + ti * 16 + ((lane & 16) ? 8 : 0) + v;
        size_t oi = (size_t)(bz * Co + co) * P + p;
        out[oi] = resid[oi] + 0.25f * accs[ti][tj][v];
      }
    }
}

// ---------------------------------------------------------------------------
extern "C" void kernel_launch(void* const* d_in, const int* in_sizes, int n_in,
                              void* d_out, int out_size, void* d_ws, size_t ws_size,
                              hipStream_t stream) {
  (void)in_sizes; (void)n_in; (void)out_size; (void)ws_size;
  const float* x      = (const float*)d_in[0];
  const float* bb0_w  = (const float*)d_in[1];
  const float* bb0_b  = (const float*)d_in[2];
  const float* bb0_a  = (const float*)d_in[3];
  const float* rb1_w1 = (const float*)d_in[4];
  const float* rb1_b1 = (const float*)d_in[5];
  const float* rb1_w2 = (const float*)d_in[6];
  const float* rb1_b2 = (const float*)d_in[7];
  const float* pamb_w = (const float*)d_in[8];
  const float* pamb_b = (const float*)d_in[9];
  const float* pamb_a = (const float*)d_in[10];
  const float* pam_w  = (const float*)d_in[11];
  const float* pam_b  = (const float*)d_in[12];
  const float* pam_a  = (const float*)d_in[13];
  const float* paa_w  = (const float*)d_in[14];
  const float* paa_b  = (const float*)d_in[15];
  const float* paa_a  = (const float*)d_in[16];
  const float* rb2_w1 = (const float*)d_in[17];
  const float* rb2_b1 = (const float*)d_in[18];
  const float* rb2_w2 = (const float*)d_in[19];
  const float* rb2_b2 = (const float*)d_in[20];
  float* out = (float*)d_out;

  const int B = 2, Ci = 64, Co = 128, C2 = 64, H = 64, W = 64, P = H * W;
  const int HS[4] = {64, 57, 51, 44};  // int(64 * {1.0, 0.9, 0.8, 0.7})
  PamGeom g; g.lofs[0] = 0;
  int ro = 0, bo = 0;
  for (int s = 0; s < 4; ++s) {
    g.Hs[s] = HS[s]; g.Ws[s] = HS[s];
    g.lofs[s + 1] = g.lofs[s] + HS[s] * HS[s];
    g.refm_off[s] = ro; g.base_off[s] = bo;
    ro += B * C2 * HS[s] * HS[s]; bo += B * Co * HS[s] * HS[s];
  }
  const int Lt  = g.lofs[4];                 // 11882
  const int Ltp = ((Lt + 31) / 32) * 32;     // 11904 (K and M padding)
  const int Kc  = Co * 9;                    // 1152
  const int K0  = Ci * 9;                    // 576

  // ---- workspace carve-out --------------------------------------------
  char* wsp = (char*)d_ws; size_t cur = 0;
  auto alloc = [&](size_t bytes) { char* p = wsp + cur; cur += (bytes + 255) & ~(size_t)255; return p; };
  auto cdiv  = [](int a, int b) { return (a + b - 1) / b; };

  float*  pool   = (float*)alloc((size_t)B * Ci * P * 4);
  float*  h0     = (float*)alloc((size_t)B * Co * P * 4);
  float*  t1     = (float*)alloc((size_t)B * Co * P * 4);
  float*  h1     = (float*)alloc((size_t)B * Co * P * 4);
  float*  mb     = (float*)alloc((size_t)B * C2 * P * 4);
  float*  pam    = (float*)alloc((size_t)B * Co * P * 4);
  float*  t2     = (float*)alloc((size_t)B * Co * P * 4);
  __bf16* imc    = (__bf16*)alloc((size_t)B * P * Kc * 2);       // im2col scratch
  // bf16 weights
  __bf16* wbb0 = (__bf16*)alloc((size_t)Co * K0 * 2);
  __bf16* wr11 = (__bf16*)alloc((size_t)Co * Kc * 2);
  __bf16* wr12 = (__bf16*)alloc((size_t)Co * Kc * 2);
  __bf16* wr21 = (__bf16*)alloc((size_t)Co * Kc * 2);
  __bf16* wr22 = (__bf16*)alloc((size_t)Co * Kc * 2);
  __bf16* wmbw = (__bf16*)alloc((size_t)C2 * Co * 2);
  __bf16* wmw  = (__bf16*)alloc((size_t)C2 * Co * 2);
  __bf16* waw  = (__bf16*)alloc((size_t)Co * Co * 2);
  // pyramid
  float*  refs[4]; refs[0] = h1;
  for (int s = 1; s < 4; ++s) refs[s] = (float*)alloc((size_t)B * Co * HS[s] * HS[s] * 4);
  __bf16* nrefs[4];
  int     Mp[4];
  for (int s = 0; s < 4; ++s) {
    Mp[s] = cdiv(B * HS[s] * HS[s], 32) * 32;
    nrefs[s] = (__bf16*)alloc((size_t)Mp[s] * Co * 2);
  }
  float*  refm_all = (float*)alloc((size_t)ro * 4);
  float*  base_all = (float*)alloc((size_t)bo * 4);
  __bf16* xpT    = (__bf16*)alloc((size_t)B * P * K0 * 2);
  __bf16* wn     = (__bf16*)alloc((size_t)B * Ltp * K0 * 2);
  float*  scores = (float*)alloc((size_t)B * Lt * P * 4);
  __bf16* attT   = (__bf16*)alloc((size_t)B * P * Ltp * 2);
  __bf16* rwT    = (__bf16*)alloc((size_t)B * 9 * Co * Ltp * 2);
  __bf16* zrow   = (__bf16*)alloc((size_t)Ltp * 2);

  // ---- weight conversion (tiny, memory-bound) -------------------------
  auto cvt = [&](const float* src, __bf16* dst, int n) {
    k_f2bf<<<cdiv(n, 256), 256, 0, stream>>>(src, dst, n);
  };
  cvt(bb0_w, wbb0, Co * K0);
  cvt(rb1_w1, wr11, Co * Kc); cvt(rb1_w2, wr12, Co * Kc);
  cvt(rb2_w1, wr21, Co * Kc); cvt(rb2_w2, wr22, Co * Kc);
  cvt(pamb_w, wmbw, C2 * Co); cvt(pam_w, wmw, C2 * Co); cvt(paa_w, waw, Co * Co);
  k_zero_bf<<<cdiv(Ltp, 256), 256, 0, stream>>>(zrow, Ltp);

  const int Mconv = B * P;                        // 8192 (multiple of 32)
  dim3 gc8(cdiv(Co / 32, WPB), Mconv / 32);       // Co=128 output tiles
  dim3 gc4(cdiv(C2 / 32, WPB), Mconv / 32);       // Co=64  output tiles

  // 1. maxpool 128^2 -> 64^2
  k_maxpool2<<<cdiv(B * Ci * P, 256), 256, 0, stream>>>(x, pool, B * Ci, H, W);

  // 2. bb0 conv3x3 (64->128) + PReLU
  k_im2col3<<<(int)cdiv(Mconv * K0, 256), 256, 0, stream>>>(pool, imc, B, Ci, H, W);
  k_gemm_conv_wmma<<<gc8, TPB, 0, stream>>>(imc, wbb0, bb0_b, bb0_a, nullptr, h0,
                                            Mconv, K0, Co, P, 2);
  // 3. ResBlock 1
  k_im2col3<<<(int)cdiv(Mconv * Kc, 256), 256, 0, stream>>>(h0, imc, B, Co, H, W);
  k_gemm_conv_wmma<<<gc8, TPB, 0, stream>>>(imc, wr11, rb1_b1, nullptr, nullptr, t1,
                                            Mconv, Kc, Co, P, 1);
  k_im2col3<<<(int)cdiv(Mconv * Kc, 256), 256, 0, stream>>>(t1, imc, B, Co, H, W);
  k_gemm_conv_wmma<<<gc8, TPB, 0, stream>>>(imc, wr12, rb1_b2, nullptr, h0, h1,
                                            Mconv, Kc, Co, P, 0);

  // 4. match_base = PReLU(1x1(h1))
  k_nhwc<<<(int)cdiv(Mconv * Co, 256), 256, 0, stream>>>(h1, nrefs[0], B, Co, P);
  k_gemm_conv_wmma<<<gc4, TPB, 0, stream>>>(nrefs[0], wmbw, pamb_b, pamb_a, nullptr, mb,
                                            Mconv, Co, C2, P, 2);

  // 5. pyramid refs + per-scale 1x1 convs
  for (int s = 1; s < 4; ++s) {
    int tot = B * Co * HS[s] * HS[s];
    k_resize<<<cdiv(tot, 256), 256, 0, stream>>>(h1, refs[s], B * Co, H, W, HS[s], HS[s]);
    k_nhwc<<<cdiv(tot, 256), 256, 0, stream>>>(refs[s], nrefs[s], B, Co, HS[s] * HS[s]);
  }
  for (int s = 0; s < 4; ++s) {
    int hw = HS[s] * HS[s], M = B * hw;
    dim3 ga(cdiv(Co / 32, WPB), Mp[s] / 32);
    k_gemm_conv_wmma<<<ga, TPB, 0, stream>>>(nrefs[s], waw, paa_b, paa_a, nullptr,
                                             base_all + g.base_off[s], M, Co, Co, hw, 2);
    dim3 gm(cdiv(C2 / 32, WPB), Mp[s] / 32);
    k_gemm_conv_wmma<<<gm, TPB, 0, stream>>>(nrefs[s], wmw, pam_b, pam_a, nullptr,
                                             refm_all + g.refm_off[s], M, Co, C2, hw, 2);
  }

  // 6. score operands
  k_make_xpT<<<(int)cdiv(Mconv * K0, 256), 256, 0, stream>>>(mb, xpT, B, C2, H, W);
  k_make_wn<<<cdiv(B * Lt, 8), 256, 0, stream>>>(refm_all, wn, g, B, C2, Ltp);

  // 7. scores = 10 * wn @ xpT^T   (M=Lt, N=P, K=576)
  dim3 gsc(cdiv(P / 32, WPB), cdiv(Lt, 32), B);
  k_gemm_scores_wmma<<<gsc, TPB, 0, stream>>>(wn, xpT, scores, Lt, Ltp, P, K0, 10.0f);

  // 8. softmax over Lt -> attT (bf16, K-major, pad zeroed)
  k_softmax<<<cdiv(B * P, 256), 256, 0, stream>>>(scores, attT, B, Lt, Ltp, P);

  // 9. flipped assembly patches (tconv BT operand, pad zeroed)
  long long rwtot = (long long)B * 9 * Co * Ltp;
  k_make_rwT<<<(unsigned)((rwtot + 255) / 256), 256, 0, stream>>>(base_all, rwT, g, B, Co, Ltp);

  // 10. tconv + /4 + residual(h1)
  dim3 gtc(cdiv(Co / 32, WPB), P / 32, B);
  k_tconv_wmma<<<gtc, TPB, 0, stream>>>(attT, rwT, zrow, h1, pam, Ltp, Co, H, W);

  // 11. ResBlock 2 -> d_out
  k_im2col3<<<(int)cdiv(Mconv * Kc, 256), 256, 0, stream>>>(pam, imc, B, Co, H, W);
  k_gemm_conv_wmma<<<gc8, TPB, 0, stream>>>(imc, wr21, rb2_b1, nullptr, nullptr, t2,
                                            Mconv, Kc, Co, P, 1);
  k_im2col3<<<(int)cdiv(Mconv * Kc, 256), 256, 0, stream>>>(t2, imc, B, Co, H, W);
  k_gemm_conv_wmma<<<gc8, TPB, 0, stream>>>(imc, wr22, rb2_b2, nullptr, pam, out,
                                            Mconv, Kc, Co, P, 0);
}
